// QKVAttentionLegacy_87892210745426
// MI455X (gfx1250) — compile-verified, matrix operand order checked by reading, not run on hardware
//
#include <hip/hip_runtime.h>
#include <hip/hip_bf16.h>

typedef __attribute__((ext_vector_type(16))) _Float16 v16h;
typedef __attribute__((ext_vector_type(8)))  float    v8f;

// Problem constants (fixed by the reference harness)
#define NBATCH 4
#define NHEADS 8
#define NB     (NBATCH * NHEADS)   // 32 batch-heads
#define CH     64                  // head dim
#define TSEQ   2048                // sequence length
#define SCALE  0.125f              // 1/sqrt(64), full scale folded into Q

#define QTILE  128                 // query rows per block (16 per wave, 8 waves)
#define KTILE  64                  // key/value rows per inner iteration
#define ST     72                  // LDS stride (halves) for K/V tiles (padded)
#define PST    72                  // LDS stride (halves) for P scratch

// A-fragment (and assumed B-fragment) half-index -> K offset within a 32-wide chunk.
// ISA 7.12.2, 16-bit A 16x32: VGPR v holds K = {2v,2v+1} (+16 for v>=4), +8 for hi half-wave.
__device__ __forceinline__ int klocal(int h, int hi) {
    int v = h >> 1, pos = h & 1;
    return ((v & 4) ? 16 : 0) + (hi ? 8 : 0) + ((v & 3) << 1) + pos;
}

__global__ __launch_bounds__(256)
void flash_attn_wmma_kernel(const float* __restrict__ q,
                            const float* __restrict__ kv,
                            float* __restrict__ out) {
    __shared__ _Float16 Ks[KTILE * ST];           // K tile: [s][c]
    __shared__ _Float16 Vs[KTILE * ST];           // V tile: [s][c]
    __shared__ _Float16 Ps[8 * 16 * PST];         // per-wave P tiles: [wave][m][s]

    const int tid  = threadIdx.x;
    const int lane = tid & 31;
    const int wave = tid >> 5;
    const int ln   = lane & 15;
    const int hi   = lane >> 4;                   // 0: lanes 0-15, 1: lanes 16-31

    const int b    = blockIdx.y;                  // batch-head 0..31
    const int t0   = blockIdx.x * QTILE + wave * 16;  // this wave's first query row

    const float* qb = q  + (size_t)b * CH * TSEQ;             // [c][t]
    const float* kb = kv + (size_t)(2 * b) * CH * TSEQ;       // K: [c][t]
    const float* vb = kb + (size_t)CH * TSEQ;                 // V: [c][t]
    float*       ob = out + (size_t)b * CH * TSEQ;            // [c][t]

    _Float16* Pw = &Ps[wave * 16 * PST];

    // Per-thread staging coordinates: element j covers (c = cb + 16j, s = sq..sq+3).
    const int sq = (tid & 15) * 4;   // 0..60, 16B-aligned s offset
    const int cb = tid >> 4;         // 0..15

    // ---- Load this wave's Q tile (16 t-rows x 64 c) into two A-fragments, scaled ----
    v16h qf[2];
    #pragma unroll
    for (int kc = 0; kc < 2; ++kc) {
        #pragma unroll
        for (int h = 0; h < 16; ++h) {
            int c = kc * 32 + klocal(h, hi);
            qf[kc][h] = (_Float16)(qb[(size_t)c * TSEQ + (t0 + ln)] * SCALE);
        }
    }

    // ---- Online softmax state + output accumulators ----
    float mrun[8], lrun[8];
    #pragma unroll
    for (int r = 0; r < 8; ++r) { mrun[r] = -3.0e38f; lrun[r] = 0.0f; }
    v8f oacc[4];
    #pragma unroll
    for (int c4 = 0; c4 < 4; ++c4) oacc[c4] = (v8f){0,0,0,0,0,0,0,0};

    // ---- Software pipeline: prefetch tile 0 into registers (128-bit loads) ----
    float4 kreg[4], vreg[4];
    #pragma unroll
    for (int j = 0; j < 4; ++j) {
        size_t off = (size_t)(cb + 16 * j) * TSEQ + sq;
        kreg[j] = *(const float4*)(kb + off);
        vreg[j] = *(const float4*)(vb + off);
    }

    for (int s0 = 0; s0 < TSEQ; s0 += KTILE) {
        __syncthreads();   // previous iteration done reading Ks/Vs

        // ---- Commit prefetched tile to LDS as f16 ----
        #pragma unroll
        for (int j = 0; j < 4; ++j) {
            int c = cb + 16 * j;
            Ks[(sq + 0) * ST + c] = (_Float16)kreg[j].x;
            Ks[(sq + 1) * ST + c] = (_Float16)kreg[j].y;
            Ks[(sq + 2) * ST + c] = (_Float16)kreg[j].z;
            Ks[(sq + 3) * ST + c] = (_Float16)kreg[j].w;
            Vs[(sq + 0) * ST + c] = (_Float16)vreg[j].x;
            Vs[(sq + 1) * ST + c] = (_Float16)vreg[j].y;
            Vs[(sq + 2) * ST + c] = (_Float16)vreg[j].z;
            Vs[(sq + 3) * ST + c] = (_Float16)vreg[j].w;
        }
        __syncthreads();

        // ---- Issue next tile's global loads; they overlap the compute below ----
        if (s0 + KTILE < TSEQ) {
            #pragma unroll
            for (int j = 0; j < 4; ++j) {
                size_t off = (size_t)(cb + 16 * j) * TSEQ + (s0 + KTILE) + sq;
                kreg[j] = *(const float4*)(kb + off);
                vreg[j] = *(const float4*)(vb + off);
            }
        }

        // ---- S = Q * K^T  (16 x 64 per wave), f32 accum ----
        v8f sacc[4];
        #pragma unroll
        for (int j = 0; j < 4; ++j) sacc[j] = (v8f){0,0,0,0,0,0,0,0};

        #pragma unroll
        for (int kc = 0; kc < 2; ++kc) {
            #pragma unroll
            for (int j = 0; j < 4; ++j) {
                v16h bk;
                #pragma unroll
                for (int h = 0; h < 16; ++h) {
                    int c = kc * 32 + klocal(h, hi);    // K dim = channel
                    int s = j * 16 + ln;                // N dim = key pos
                    bk[h] = Ks[s * ST + c];
                }
                sacc[j] = __builtin_amdgcn_wmma_f32_16x16x32_f16(
                    false, qf[kc], false, bk, (short)0, sacc[j], false, false);
            }
        }

        // ---- Online softmax over this 64-wide slab; write P (f16) to LDS ----
        #pragma unroll
        for (int r = 0; r < 8; ++r) {
            float x0 = sacc[0][r], x1 = sacc[1][r], x2 = sacc[2][r], x3 = sacc[3][r];
            float mloc = fmaxf(fmaxf(x0, x1), fmaxf(x2, x3));
            #pragma unroll
            for (int mask = 1; mask <= 8; mask <<= 1)
                mloc = fmaxf(mloc, __shfl_xor(mloc, mask, 32));
            float mnew = fmaxf(mrun[r], mloc);
            float corr = __expf(mrun[r] - mnew);
            float p0 = __expf(x0 - mnew), p1 = __expf(x1 - mnew);
            float p2 = __expf(x2 - mnew), p3 = __expf(x3 - mnew);
            float ls = p0 + p1 + p2 + p3;
            #pragma unroll
            for (int mask = 1; mask <= 8; mask <<= 1)
                ls += __shfl_xor(ls, mask, 32);
            lrun[r] = lrun[r] * corr + ls;
            mrun[r] = mnew;
            #pragma unroll
            for (int c4 = 0; c4 < 4; ++c4) oacc[c4][r] *= corr;
            int m = r + (hi ? 8 : 0);
            Pw[m * PST + 0  + ln] = (_Float16)p0;
            Pw[m * PST + 16 + ln] = (_Float16)p1;
            Pw[m * PST + 32 + ln] = (_Float16)p2;
            Pw[m * PST + 48 + ln] = (_Float16)p3;
        }

        // ---- O += P * V  (16 x 64 per wave) ----
        #pragma unroll
        for (int sc = 0; sc < 2; ++sc) {
            v16h ap;
            #pragma unroll
            for (int h = 0; h < 16; ++h) {
                int s = sc * 32 + klocal(h, hi);        // K dim = key pos
                ap[h] = Pw[ln * PST + s];               // M = ln
            }
            #pragma unroll
            for (int c4 = 0; c4 < 4; ++c4) {
                v16h bv;
                #pragma unroll
                for (int h = 0; h < 16; ++h) {
                    int s = sc * 32 + klocal(h, hi);    // K dim = key pos
                    int c = c4 * 16 + ln;               // N dim = channel
                    bv[h] = Vs[s * ST + c];
                }
                oacc[c4] = __builtin_amdgcn_wmma_f32_16x16x32_f16(
                    false, ap, false, bv, (short)0, oacc[c4], false, false);
            }
        }
    }

    // ---- Normalize and write out[c][t] with 128-bit stores ----
    float inv[8];
    #pragma unroll
    for (int r = 0; r < 8; ++r) inv[r] = 1.0f / lrun[r];
    const int tb = t0 + (hi ? 8 : 0);   // rows r=0..7 -> t = tb + r (consecutive)
    #pragma unroll
    for (int c4 = 0; c4 < 4; ++c4) {
        int c = c4 * 16 + ln;
        float4 w0 = { oacc[c4][0] * inv[0], oacc[c4][1] * inv[1],
                      oacc[c4][2] * inv[2], oacc[c4][3] * inv[3] };
        float4 w1 = { oacc[c4][4] * inv[4], oacc[c4][5] * inv[5],
                      oacc[c4][6] * inv[6], oacc[c4][7] * inv[7] };
        *(float4*)(ob + (size_t)c * TSEQ + tb)     = w0;
        *(float4*)(ob + (size_t)c * TSEQ + tb + 4) = w1;
    }
}

extern "C" void kernel_launch(void* const* d_in, const int* in_sizes, int n_in,
                              void* d_out, int out_size, void* d_ws, size_t ws_size,
                              hipStream_t stream) {
    const float* q  = (const float*)d_in[0];
    const float* kv = (const float*)d_in[1];
    // d_in[2] = n_heads (scalar); shapes are compile-time constants here.
    float* out = (float*)d_out;

    dim3 grid(TSEQ / QTILE, NB);   // (16, 32)
    dim3 block(256);
    flash_attn_wmma_kernel<<<grid, block, 0, stream>>>(q, kv, out);
}
